// HeteroSAGE_60619168416014
// MI455X (gfx1250) — matrix operand order repaired and checked by reading, not compiled, not run on hardware
//
#include <hip/hip_runtime.h>

// ---------------------------------------------------------------------------
// HeteroSAGE on MI455X (gfx1250), wave32 + WMMA f32_16x16x32_f16.
//   0) memsetAsync zero of accumulators (agg, cnt, pool, pcnt)
//   1) edge_scatter  x6  : atomic feature scatter-add + degree count
//   2) to_h16_t      x16 : f32 -> f16 TRANSPOSED weight staging ([N][K] so
//                          B-fragments are contiguous 16B loads)
//   3) seg_count     x5  : per-graph node counts
//   4) sage_node_pool x5 : FUSED: coalesced LDS tile staging -> WMMA SAGE
//                          linears -> bias+ReLU -> WMMA node linear ->
//                          pooled atomic accumulation. One wave / 16 nodes.
//   5) final_out          : [G,640]x[640,64] output projection.
// ---------------------------------------------------------------------------

typedef _Float16 half_t;
typedef __attribute__((ext_vector_type(16))) _Float16 v16h;
typedef __attribute__((ext_vector_type(8)))  _Float16 v8h;
typedef __attribute__((ext_vector_type(8)))  float    v8f;

#define HID 128
#define OUTD 64

// ---- WMMA fragment builders (CDNA5 ISA 7.12.2 layouts, wave32) ------------

// A-matrix 16x32 f16 from an LDS-staged row-major fp32 tile (tail rows were
// zero-filled at staging time -> branch-free). k runs are contiguous float4s.
__device__ __forceinline__ v16h a_frag_sm(const float* __restrict__ sm, int ld,
                                          int kbase, int lane, float scale) {
  const int row = lane & 15;
  const int khi = (lane >> 4) << 3;            // 0 or 8
  const float* rp = sm + row * ld + kbase + khi;
  const float4 f0 = *(const float4*)(rp + 0);   // k: +0..3
  const float4 f1 = *(const float4*)(rp + 4);   // k: +4..7
  const float4 f2 = *(const float4*)(rp + 16);  // k: +16..19
  const float4 f3 = *(const float4*)(rp + 20);  // k: +20..23
  v16h a;
  a[0]  = (half_t)(f0.x * scale); a[1]  = (half_t)(f0.y * scale);
  a[2]  = (half_t)(f0.z * scale); a[3]  = (half_t)(f0.w * scale);
  a[4]  = (half_t)(f1.x * scale); a[5]  = (half_t)(f1.y * scale);
  a[6]  = (half_t)(f1.z * scale); a[7]  = (half_t)(f1.w * scale);
  a[8]  = (half_t)(f2.x * scale); a[9]  = (half_t)(f2.y * scale);
  a[10] = (half_t)(f2.z * scale); a[11] = (half_t)(f2.w * scale);
  a[12] = (half_t)(f3.x * scale); a[13] = (half_t)(f3.y * scale);
  a[14] = (half_t)(f3.z * scale); a[15] = (half_t)(f3.w * scale);
  return a;
}

// A-matrix 16x32 f16 from the LDS-resident row-major f16 h tile.
__device__ __forceinline__ v16h a_frag_h(const half_t* __restrict__ tile,
                                         int kbase, int lane) {
  const int row = lane & 15;
  const int khi = (lane >> 4) << 3;
  const half_t* rp = tile + row * HID + kbase + khi;
  const v8h lo = *(const v8h*)rp;          // k: +0..7
  const v8h hi = *(const v8h*)(rp + 16);   // k: +16..23
  v16h a;
#pragma unroll
  for (int j = 0; j < 8; ++j) { a[j] = lo[j]; a[j + 8] = hi[j]; }
  return a;
}

// B-matrix 32x16 f16 from TRANSPOSED f16 weights WT[N][K]: per lane two
// 16B loads, k: +0..7 and +16..23 (table order, matches A).
__device__ __forceinline__ v16h b_frag(const half_t* __restrict__ WT, int K,
                                       int kbase, int nbase, int lane) {
  const int col = nbase + (lane & 15);
  const int k0 = kbase + ((lane >> 4) << 3);   // 0 or 8 within chunk
  const half_t* bp = WT + (size_t)col * K + k0;
  const v8h lo = *(const v8h*)bp;
  const v8h hi = *(const v8h*)(bp + 16);
  return __builtin_shufflevector(lo, hi, 0, 1, 2, 3, 4, 5, 6, 7, 8, 9, 10, 11, 12, 13, 14, 15);
}

// ---- Kernels ---------------------------------------------------------------

__global__ __launch_bounds__(256) void edge_scatter(
    const float* __restrict__ xsrc, const int* __restrict__ src,
    const int* __restrict__ dst, long long total, int logD,
    float* __restrict__ agg, float* __restrict__ cnt) {
  const long long idx = (long long)blockIdx.x * blockDim.x + threadIdx.x;
  if (idx >= total) return;
  const int D = 1 << logD;
  const long long e = idx >> logD;
  const int f = (int)(idx & (D - 1));
  const int s = src[e];
  const int d = dst[e];
  atomicAdd(&agg[((size_t)d << logD) + f], xsrc[((size_t)s << logD) + f]);
  if (f == 0) atomicAdd(&cnt[d], 1.0f);
}

// f32 -> f16 with transpose: in is row-major [K][HID], out is [HID][K].
__global__ __launch_bounds__(256) void to_h16_t(const float* __restrict__ a,
                                                const float* __restrict__ b,
                                                half_t* __restrict__ o,
                                                int K, int n) {
  const int i = blockIdx.x * blockDim.x + threadIdx.x;
  if (i >= n) return;
  const int k = i / HID;
  const int c = i - k * HID;
  float v = a[i];
  if (b) v += b[i];
  o[(size_t)c * K + k] = (half_t)v;
}

__global__ __launch_bounds__(256) void seg_count(const int* __restrict__ batch,
                                                 int n, float* __restrict__ pcnt) {
  const int i = blockIdx.x * blockDim.x + threadIdx.x;
  if (i < n) atomicAdd(&pcnt[batch[i]], 1.0f);
}

// Fused SAGE + node-linear + pooled accumulation. One wave32 per 16-node tile.
__global__ __launch_bounds__(256) void sage_node_pool(
    int N,
    const float* __restrict__ xdst, int Ddst, const half_t* __restrict__ WrT,
    const float* __restrict__ agg0, const float* __restrict__ cnt0,
    const half_t* __restrict__ Wl0T, int Dsrc0, const float* __restrict__ bl0,
    const float* __restrict__ agg1, const float* __restrict__ cnt1,
    const half_t* __restrict__ Wl1T, int Dsrc1, const float* __restrict__ bl1,
    const half_t* __restrict__ nWT, const float* __restrict__ nb,
    const int* __restrict__ batch, float* __restrict__ pool) {
  // 4KB per wave: fp32 16x64 A-tile staging, later reused as f16 16x128 h tile.
  __shared__ __align__(16) float smem[8][16 * 64];
  const int wave = threadIdx.x >> 5;
  const int lane = threadIdx.x & 31;
  const int t0 = (blockIdx.x * 8 + wave) * 16;   // tile id uniform per wave
  if (t0 >= N) return;                            // whole-wave exit: EXEC full
  const int rows = min(16, N - t0);
  const int row = lane & 15;
  const int col = lane & 15;
  const int hi  = lane >> 4;
  float* sm = &smem[wave][0];

  // Coalesced cooperative staging of a 16 x D fp32 tile (zero-fill tail rows).
  auto stage = [&](const float* __restrict__ gbase, int ld) {
    const int l4 = ld >> 2;                      // float4 per row: 8 or 16
    const float4* g4 = (const float4*)gbase;
    float4* s4 = (float4*)sm;
    const int nvec = l4 << 4;                    // 16 rows
    for (int i = lane; i < nvec; i += 32) {
      const int r = (l4 == 16) ? (i >> 4) : (i >> 3);
      float4 v = {0.0f, 0.0f, 0.0f, 0.0f};
      if (r < rows) v = g4[i];
      s4[i] = v;
    }
  };

  v8f acc[8] = {};                                // 16x128 fp32 C accumulator

  // x_dst @ Wr
  stage(xdst + (size_t)t0 * Ddst, Ddst);
  for (int kc = 0; kc < Ddst; kc += 32) {
    v16h a = a_frag_sm(sm, Ddst, kc, lane, 1.0f);
#pragma unroll
    for (int n = 0; n < 8; ++n) {
      v16h b = b_frag(WrT, Ddst, kc, n * 16, lane);
      acc[n] = __builtin_amdgcn_wmma_f32_16x16x32_f16(
          false, a, false, b, (short)0, acc[n], false, false);
    }
  }
  // mean(agg0) @ Wl0
  {
    const float c = (row < rows) ? cnt0[t0 + row] : 1.0f;
    const float inv = 1.0f / fmaxf(c, 1.0f);
    stage(agg0 + (size_t)t0 * Dsrc0, Dsrc0);
    for (int kc = 0; kc < Dsrc0; kc += 32) {
      v16h a = a_frag_sm(sm, Dsrc0, kc, lane, inv);
#pragma unroll
      for (int n = 0; n < 8; ++n) {
        v16h b = b_frag(Wl0T, Dsrc0, kc, n * 16, lane);
        acc[n] = __builtin_amdgcn_wmma_f32_16x16x32_f16(
            false, a, false, b, (short)0, acc[n], false, false);
      }
    }
  }
  // optional second conv branch (instruction: 'ii'; Wr_vi+Wr_ii pre-summed)
  if (Wl1T) {
    const float c = (row < rows) ? cnt1[t0 + row] : 1.0f;
    const float inv = 1.0f / fmaxf(c, 1.0f);
    stage(agg1 + (size_t)t0 * Dsrc1, Dsrc1);
    for (int kc = 0; kc < Dsrc1; kc += 32) {
      v16h a = a_frag_sm(sm, Dsrc1, kc, lane, inv);
#pragma unroll
      for (int n = 0; n < 8; ++n) {
        v16h b = b_frag(Wl1T, Dsrc1, kc, n * 16, lane);
        acc[n] = __builtin_amdgcn_wmma_f32_16x16x32_f16(
            false, a, false, b, (short)0, acc[n], false, false);
      }
    }
  }

  // bias + ReLU -> f16 h tile in LDS (elem acc[n][r] = H[r+8*hi][16n+col]).
  half_t* hb = (half_t*)sm;                       // reuse the staging buffer
#pragma unroll
  for (int n = 0; n < 8; ++n) {
    const float bb = bl0[n * 16 + col] + (bl1 ? bl1[n * 16 + col] : 0.0f);
#pragma unroll
    for (int r = 0; r < 8; ++r) {
      const int m = r + hi * 8;
      hb[m * HID + n * 16 + col] = (half_t)fmaxf(acc[n][r] + bb, 0.0f);
    }
  }

  // z = h @ nW  (128x128: 4 K-chunks x 8 N-tiles of WMMA)
  v8f z[8] = {};
#pragma unroll
  for (int kc = 0; kc < HID; kc += 32) {
    v16h a = a_frag_h(hb, kc, lane);
#pragma unroll
    for (int n = 0; n < 8; ++n) {
      v16h b = b_frag(nWT, HID, kc, n * 16, lane);
      z[n] = __builtin_amdgcn_wmma_f32_16x16x32_f16(
          false, a, false, b, (short)0, z[n], false, false);
    }
  }

  // Pooled accumulation: row m -> graph batch[t0+m]; adding nb once per node
  // is exact (segment-mean divides by the same count).
#pragma unroll
  for (int r = 0; r < 8; ++r) {
    const int m = r + hi * 8;
    if (m < rows) {
      const int g = batch[t0 + m];
      float* pg = pool + (size_t)g * HID;
#pragma unroll
      for (int n = 0; n < 8; ++n) {
        atomicAdd(&pg[n * 16 + col], z[n][r] + nb[n * 16 + col]);
      }
    }
  }
}

__global__ __launch_bounds__(64) void final_out(
    const float* __restrict__ pool, const float* __restrict__ pcnt,
    const float* __restrict__ gW, const float* __restrict__ gb,
    float* __restrict__ out, int Gn) {
  const int g = blockIdx.x;
  const int o = threadIdx.x;
  float s = gb[o];
#pragma unroll
  for (int t = 0; t < 5; ++t) {
    const float inv = 1.0f / fmaxf(pcnt[(size_t)t * Gn + g], 1.0f);
    const float* pp = pool + ((size_t)t * Gn + g) * HID;
    for (int c = 0; c < HID; ++c) {
      s += pp[c] * inv * gW[(size_t)(t * HID + c) * OUTD + o];
    }
  }
  out[(size_t)g * OUTD + o] = s;
}

// ---------------------------------------------------------------------------

extern "C" void kernel_launch(void* const* d_in, const int* in_sizes, int n_in,
                              void* d_out, int out_size, void* d_ws, size_t ws_size,
                              hipStream_t stream) {
  constexpr int NV = 500000, NI = 300000, NT = 50000, NA = 50000, NS = 10000;
  constexpr int Gn = 2048;
  constexpr long long EVT = 500000, EIV = 1500000, EVI = 1500000,
                      EVA = 500000, ETS = 50000, EII = 600000;

  const float* x_value       = (const float*)d_in[0];
  const float* x_typ         = (const float*)d_in[1];
  const float* x_size        = (const float*)d_in[2];
  const float* x_attribute   = (const float*)d_in[3];
  const float* x_instruction = (const float*)d_in[4];
  const float* Wl_vt = (const float*)d_in[5];  const float* bl_vt = (const float*)d_in[6];  const float* Wr_vt = (const float*)d_in[7];
  const float* Wl_iv = (const float*)d_in[8];  const float* bl_iv = (const float*)d_in[9];  const float* Wr_iv = (const float*)d_in[10];
  const float* Wl_vi = (const float*)d_in[11]; const float* bl_vi = (const float*)d_in[12]; const float* Wr_vi = (const float*)d_in[13];
  const float* Wl_va = (const float*)d_in[14]; const float* bl_va = (const float*)d_in[15]; const float* Wr_va = (const float*)d_in[16];
  const float* Wl_ts = (const float*)d_in[17]; const float* bl_ts = (const float*)d_in[18]; const float* Wr_ts = (const float*)d_in[19];
  const float* Wl_ii = (const float*)d_in[20]; const float* bl_ii = (const float*)d_in[21]; const float* Wr_ii = (const float*)d_in[22];
  const float* nW_typ = (const float*)d_in[23];         const float* nb_typ = (const float*)d_in[24];
  const float* nW_value = (const float*)d_in[25];       const float* nb_value = (const float*)d_in[26];
  const float* nW_instruction = (const float*)d_in[27]; const float* nb_instruction = (const float*)d_in[28];
  const float* nW_attribute = (const float*)d_in[29];   const float* nb_attribute = (const float*)d_in[30];
  const float* nW_size = (const float*)d_in[31];        const float* nb_size = (const float*)d_in[32];
  const float* gW = (const float*)d_in[33];             const float* gb = (const float*)d_in[34];
  const int* src_vt = (const int*)d_in[35]; const int* dst_vt = (const int*)d_in[36];
  const int* src_iv = (const int*)d_in[37]; const int* dst_iv = (const int*)d_in[38];
  const int* src_vi = (const int*)d_in[39]; const int* dst_vi = (const int*)d_in[40];
  const int* src_va = (const int*)d_in[41]; const int* dst_va = (const int*)d_in[42];
  const int* src_ts = (const int*)d_in[43]; const int* dst_ts = (const int*)d_in[44];
  const int* src_ii = (const int*)d_in[45]; const int* dst_ii = (const int*)d_in[46];
  const int* batch_typ = (const int*)d_in[47];
  const int* batch_value = (const int*)d_in[48];
  const int* batch_instruction = (const int*)d_in[49];
  const int* batch_attribute = (const int*)d_in[50];
  const int* batch_size = (const int*)d_in[51];

  // ---- workspace layout ----
  float* fws = (float*)d_ws;
  size_t fo = 0;
  auto AF = [&](size_t n) { float* p = fws + fo; fo += n; return p; };
  float* agg_vt = AF((size_t)NT * 64); float* cnt_vt = AF(NT);
  float* agg_iv = AF((size_t)NV * 64); float* cnt_iv = AF(NV);
  float* agg_vi = AF((size_t)NI * 64); float* cnt_vi = AF(NI);
  float* agg_ii = AF((size_t)NI * 64); float* cnt_ii = AF(NI);
  float* agg_va = AF((size_t)NA * 64); float* cnt_va = AF(NA);
  float* agg_ts = AF((size_t)NS * 32); float* cnt_ts = AF(NS);
  float* pool = AF((size_t)5 * Gn * HID);
  float* pcnt = AF((size_t)5 * Gn);
  const size_t zero_bytes = fo * sizeof(float);

  half_t* hws = (half_t*)(fws + fo);
  size_t hoo = 0;
  auto AH = [&](size_t n) { half_t* p = hws + hoo; hoo += n; return p; };
  half_t* hWl_vt = AH(64 * HID); half_t* hWr_vt = AH(32 * HID);
  half_t* hWl_iv = AH(64 * HID); half_t* hWr_iv = AH(64 * HID);
  half_t* hWl_vi = AH(64 * HID); half_t* hWl_ii = AH(64 * HID);
  half_t* hWr_vi_ii = AH(64 * HID);              // Wr_vi + Wr_ii (same x_dst)
  half_t* hWl_va = AH(64 * HID); half_t* hWr_va = AH(32 * HID);
  half_t* hWl_ts = AH(32 * HID); half_t* hWr_ts = AH(32 * HID);
  half_t* hnW_typ = AH(HID * HID); half_t* hnW_value = AH(HID * HID);
  half_t* hnW_instruction = AH(HID * HID);
  half_t* hnW_attribute = AH(HID * HID); half_t* hnW_size = AH(HID * HID);

  hipMemsetAsync(d_ws, 0, zero_bytes, stream);

  // ---- weight staging (f32 -> f16, transposed to [N][K]) ----
  auto cvt = [&](const float* a, const float* b, half_t* o, int K) {
    const int n = K * HID;
    to_h16_t<<<(n + 255) / 256, 256, 0, stream>>>(a, b, o, K, n);
  };
  cvt(Wl_vt, nullptr, hWl_vt, 64); cvt(Wr_vt, nullptr, hWr_vt, 32);
  cvt(Wl_iv, nullptr, hWl_iv, 64); cvt(Wr_iv, nullptr, hWr_iv, 64);
  cvt(Wl_vi, nullptr, hWl_vi, 64); cvt(Wl_ii, nullptr, hWl_ii, 64);
  cvt(Wr_vi, Wr_ii, hWr_vi_ii, 64);
  cvt(Wl_va, nullptr, hWl_va, 64); cvt(Wr_va, nullptr, hWr_va, 32);
  cvt(Wl_ts, nullptr, hWl_ts, 32); cvt(Wr_ts, nullptr, hWr_ts, 32);
  cvt(nW_typ, nullptr, hnW_typ, HID);
  cvt(nW_value, nullptr, hnW_value, HID);
  cvt(nW_instruction, nullptr, hnW_instruction, HID);
  cvt(nW_attribute, nullptr, hnW_attribute, HID);
  cvt(nW_size, nullptr, hnW_size, HID);

  // ---- edge scatter (mean numerators + degrees) ----
  auto scat = [&](const float* xs, const int* s, const int* d, long long E,
                  int logD, float* agg, float* cnt) {
    const long long tot = E << logD;
    edge_scatter<<<(int)((tot + 255) / 256), 256, 0, stream>>>(xs, s, d, tot, logD, agg, cnt);
  };
  scat(x_value,       src_vt, dst_vt, EVT, 6, agg_vt, cnt_vt);
  scat(x_instruction, src_iv, dst_iv, EIV, 6, agg_iv, cnt_iv);
  scat(x_value,       src_vi, dst_vi, EVI, 6, agg_vi, cnt_vi);
  scat(x_instruction, src_ii, dst_ii, EII, 6, agg_ii, cnt_ii);
  scat(x_value,       src_va, dst_va, EVA, 6, agg_va, cnt_va);
  scat(x_typ,         src_ts, dst_ts, ETS, 5, agg_ts, cnt_ts);

  // ---- per-graph node counts ----
  seg_count<<<(NT + 255) / 256, 256, 0, stream>>>(batch_typ, NT, pcnt + 0 * (size_t)Gn);
  seg_count<<<(NV + 255) / 256, 256, 0, stream>>>(batch_value, NV, pcnt + 1 * (size_t)Gn);
  seg_count<<<(NI + 255) / 256, 256, 0, stream>>>(batch_instruction, NI, pcnt + 2 * (size_t)Gn);
  seg_count<<<(NA + 255) / 256, 256, 0, stream>>>(batch_attribute, NA, pcnt + 3 * (size_t)Gn);
  seg_count<<<(NS + 255) / 256, 256, 0, stream>>>(batch_size, NS, pcnt + 4 * (size_t)Gn);

  // ---- fused SAGE + node-linear + pooling ----
  auto fuse = [&](int N, const float* xd, int Dd, const half_t* WrT,
                  const float* a0, const float* c0, const half_t* W0, int D0, const float* b0,
                  const float* a1, const float* c1, const half_t* W1, int D1, const float* b1,
                  const half_t* nWT, const float* nbp, const int* batch, float* pl) {
    const int tiles = (N + 15) / 16;
    sage_node_pool<<<(tiles + 7) / 8, 256, 0, stream>>>(
        N, xd, Dd, WrT, a0, c0, W0, D0, b0, a1, c1, W1, D1, b1, nWT, nbp, batch, pl);
  };
  fuse(NT, x_typ, 32, hWr_vt, agg_vt, cnt_vt, hWl_vt, 64, bl_vt,
       nullptr, nullptr, nullptr, 0, nullptr,
       hnW_typ, nb_typ, batch_typ, pool + 0 * (size_t)Gn * HID);
  fuse(NV, x_value, 64, hWr_iv, agg_iv, cnt_iv, hWl_iv, 64, bl_iv,
       nullptr, nullptr, nullptr, 0, nullptr,
       hnW_value, nb_value, batch_value, pool + 1 * (size_t)Gn * HID);
  fuse(NI, x_instruction, 64, hWr_vi_ii, agg_vi, cnt_vi, hWl_vi, 64, bl_vi,
       agg_ii, cnt_ii, hWl_ii, 64, bl_ii,
       hnW_instruction, nb_instruction, batch_instruction, pool + 2 * (size_t)Gn * HID);
  fuse(NA, x_attribute, 32, hWr_va, agg_va, cnt_va, hWl_va, 64, bl_va,
       nullptr, nullptr, nullptr, 0, nullptr,
       hnW_attribute, nb_attribute, batch_attribute, pool + 3 * (size_t)Gn * HID);
  fuse(NS, x_size, 32, hWr_ts, agg_ts, cnt_ts, hWl_ts, 32, bl_ts,
       nullptr, nullptr, nullptr, 0, nullptr,
       hnW_size, nb_size, batch_size, pool + 4 * (size_t)Gn * HID);

  // ---- final projection ----
  final_out<<<Gn, OUTD, 0, stream>>>(pool, pcnt, gW, gb, (float*)d_out, Gn);
}
